// MetaRNN_24584392802760
// MI455X (gfx1250) — matrix operand order repaired and checked by reading, not compiled, not run on hardware
//
#include <hip/hip_runtime.h>
#include <hip/hip_bf16.h>

// ---------------------------------------------------------------------------
// MetaRNN on gfx1250 (MI455X), bf16 WMMA pipeline:
//   k_transpose_bf16 : W_ih/W_hh/W_out -> N-major bf16 (B-fragment friendly)
//   k_inproj         : xW[t,:,b] = x[b,t,:] @ W_ih + b_h  (LDS-staged A, WMMA)
//   k_scan           : persistent single-WGP recurrence, ping-pong h in LDS
//   k_outproj        : out[b,t,:] = hs[b,t,:] @ W_out + b_out
// WMMA inner loops are 2-stage software-pipelined and pinned with
// sched_group_barrier (loads of k+1 issue before WMMAs of k) so the waits in
// front of the matrix ops are partial instead of s_wait_loadcnt 0x0.
// ---------------------------------------------------------------------------

#define B_    32
#define T_    4096
#define DIN_  256
#define DH_   512
#define DOUT_ 256

// sched_group_barrier masks: 0x8 MFMA/WMMA, 0x20 VMEM read, 0x100 DS read
#if defined(__has_builtin)
#  if __has_builtin(__builtin_amdgcn_sched_group_barrier)
#    define SCHED_GB(mask, n) __builtin_amdgcn_sched_group_barrier((mask), (n), 0)
#  endif
#endif
#ifndef SCHED_GB
#  define SCHED_GB(mask, n)
#endif

typedef __attribute__((ext_vector_type(16))) __bf16 v16bf;
typedef __attribute__((ext_vector_type(8)))  __bf16 v8bf;
typedef __attribute__((ext_vector_type(8)))  float  v8f;

static __device__ inline unsigned short bfbits(float f) {
    return __builtin_bit_cast(unsigned short, (__bf16)f);   // RNE
}

// 16x32 bf16 A-fragment from a row-major bf16 row pointer.  contiguous-k layout (B loads match):
//   lanes 0-15 : M=lane,    K = kk..kk+15
//   lanes16-31 : M=lane-16, K = kk+16..kk+31
// Caller passes ka = kk + half*16.
static __device__ inline v16bf load_a_bf16(const unsigned short* rowbase, int ka) {
    v8bf lo = *(const v8bf*)(rowbase + ka);
    v8bf hi = *(const v8bf*)(rowbase + ka + 8);
    v16bf a;
#pragma unroll
    for (int i = 0; i < 8; ++i) { a[i] = lo[i]; a[8 + i] = hi[i]; }
    return a;
}

// ---------------------------------------------------------------------------
// Weight transpose + f32 -> bf16.  src [rows][cols] -> dst [cols][rows].
// ---------------------------------------------------------------------------
__global__ void k_transpose_bf16(const float* __restrict__ src,
                                 unsigned short* __restrict__ dst,
                                 int rows, int cols) {
    int i = blockIdx.x * blockDim.x + threadIdx.x;
    if (i >= rows * cols) return;
    int r = i / cols, c = i - r * cols;
    dst[(size_t)c * rows + r] = bfbits(src[i]);
}

// ---------------------------------------------------------------------------
// Input projection: xW = x @ W_ih + b_h, written scan-friendly [T][DH][B] f32.
// One block per 16-row M-tile; A staged (f32->bf16) once into LDS, each of the
// 8 waves computes 4 N-tiles sharing the staged A (x read exactly once).
// ---------------------------------------------------------------------------
__global__ __launch_bounds__(256, 1) void k_inproj(const float* __restrict__ x,
                                                   const unsigned short* __restrict__ wt_ih, // [DH][DIN] bf16
                                                   const float* __restrict__ bh,
                                                   float* __restrict__ xw) {
    __shared__ unsigned short xa[16 * DIN_];             // 8 KB staged A tile
    const int tid = threadIdx.x;
    const int m0  = blockIdx.x << 4;                     // 8192 M-tiles

    for (int i = tid; i < 16 * DIN_; i += 256) {         // stage + convert once
        int r = i >> 8, c = i & (DIN_ - 1);
        xa[i] = bfbits(x[(size_t)(m0 + r) * DIN_ + c]);
    }
    __syncthreads();

    const int lane = tid & 31, wave = tid >> 5;
    const int lrow = lane & 15, half = lane >> 4;
    const unsigned short* arow = xa + lrow * DIN_;

    const unsigned short* brow[4];
    int n0[4];
#pragma unroll
    for (int q = 0; q < 4; ++q) {
        n0[q]   = ((wave << 2) + q) << 4;                // waves cover all 32 N-tiles
        brow[q] = wt_ih + (size_t)(n0[q] + lrow) * DIN_;
    }

    v8f c[4] = {};
    // 2-stage pipeline over K
    v16bf pa = load_a_bf16(arow, half * 16);
    v16bf pb[4];
#pragma unroll
    for (int q = 0; q < 4; ++q) pb[q] = *(const v16bf*)(brow[q] + half * 16);

#pragma unroll
    for (int kk = 0; kk < DIN_; kk += 32) {
        const int kn = (kk + 32 < DIN_) ? kk + 32 : kk;  // clamp on last iter
        v16bf na = load_a_bf16(arow, kn + half * 16);
        v16bf nb[4];
#pragma unroll
        for (int q = 0; q < 4; ++q) nb[q] = *(const v16bf*)(brow[q] + kn + half * 16);

#pragma unroll
        for (int q = 0; q < 4; ++q)
            c[q] = __builtin_amdgcn_wmma_f32_16x16x32_bf16(false, pa, false, pb[q],
                                                           (short)0, c[q], false, false);
        pa = na;
#pragma unroll
        for (int q = 0; q < 4; ++q) pb[q] = nb[q];

        // per-iteration schedule: k+1 loads first, then k's WMMAs
        SCHED_GB(0x020, 8);   // 8 global_load_b128 (4 B-fragments)
        SCHED_GB(0x100, 2);   // 2 ds_load_b128 (A-fragment)
        SCHED_GB(0x008, 4);   // 4 v_wmma
    }

#pragma unroll
    for (int q = 0; q < 4; ++q) {
        const int col  = n0[q] + lrow;
        const float bias = bh[col];
#pragma unroll
        for (int j = 0; j < 8; ++j) {                    // C: lane<16 -> M=j, else M=j+8
            const int m  = m0 + half * 8 + j;
            const int b_ = m >> 12;                      // m / T
            const int t  = m & (T_ - 1);
            xw[((size_t)t * DH_ + col) * B_ + b_] = c[q][j] + bias;
        }
    }
}

// ---------------------------------------------------------------------------
// Recurrent scan: single persistent workgroup (1024 thr = 32 waves) on one WGP.
// Ping-pong h buffers in LDS (2 x 32 KB bf16) -> ONE barrier per step.
// W_hh^T (512 KB bf16) streamed from L2 each step; xW chunk for the step is
// preloaded as two v8f before the K-loop (latency hidden under 32 WMMAs) and
// the next step's chunk is prefetched (global_prefetch_b8).
// ---------------------------------------------------------------------------
__global__ __launch_bounds__(1024) void k_scan(const float* __restrict__ h0,
                                               const unsigned short* __restrict__ wt_hh, // [DH][DH] bf16
                                               const float* __restrict__ xw,             // [T][DH][B] f32
                                               unsigned short* __restrict__ hs) {        // [B][T][DH] bf16
    __shared__ unsigned short hbuf[2][B_ * DH_];         // 2 x 32 KB
    const int tid = threadIdx.x;

    for (int i = tid; i < B_ * DH_; i += 1024)           // h <- h0 (f32 -> bf16)
        hbuf[0][i] = bfbits(h0[i]);
    __syncthreads();

    const int lane = tid & 31;
    const int w    = tid >> 5;                           // 0..31 -> N tile
    const int lrow = lane & 15, half = lane >> 4;
    const int col  = (w << 4) + lrow;
    const unsigned short* brow = wt_hh + (size_t)col * DH_;

    int p = 0;
    for (int t = 0; t < T_; ++t) {
        const float* xl = xw + ((size_t)t * DH_ + col) * B_;
        __builtin_prefetch(xl + (size_t)DH_ * B_, 0, 1); // next step's chunk
        v8f xv0 = *(const v8f*)(xl + half * 8);          // rows 0-15 addends
        v8f xv1 = *(const v8f*)(xl + 16 + half * 8);     // rows 16-31 addends

        const unsigned short* hcur = hbuf[p];
        unsigned short*       hnxt = hbuf[p ^ 1];
        const unsigned short* ar0  = hcur + (size_t)lrow * DH_;         // rows 0-15
        const unsigned short* ar1  = hcur + (size_t)(16 + lrow) * DH_;  // rows 16-31

        v8f c0 = {}, c1 = {};
        // 2-stage pipeline over K
        v16bf pa0 = load_a_bf16(ar0, half * 16);
        v16bf pa1 = load_a_bf16(ar1, half * 16);
        v16bf pb  = *(const v16bf*)(brow + half * 16);

#pragma unroll 4
        for (int kk = 0; kk < DH_; kk += 32) {
            const int kn = (kk + 32 < DH_) ? kk + 32 : kk;
            const int ka = kn + half * 16;
            v16bf na0 = load_a_bf16(ar0, ka);
            v16bf na1 = load_a_bf16(ar1, ka);
            v16bf nb  = *(const v16bf*)(brow + kn + half * 16);

            c0 = __builtin_amdgcn_wmma_f32_16x16x32_bf16(false, pa0, false, pb,
                                                         (short)0, c0, false, false);
            c1 = __builtin_amdgcn_wmma_f32_16x16x32_bf16(false, pa1, false, pb,
                                                         (short)0, c1, false, false);
            pa0 = na0; pa1 = na1; pb = nb;

            // per-iteration schedule: k+1 loads first, then k's WMMAs
            SCHED_GB(0x020, 2);   // 2 global_load_b128 (B-fragment)
            SCHED_GB(0x100, 4);   // 4 ds_load_b128 (A0/A1 fragments)
            SCHED_GB(0x008, 2);   // 2 v_wmma
        }

#pragma unroll
        for (int j = 0; j < 8; ++j) {
            const int r0 = half * 8 + j;                 // batch rows 0-15
            unsigned short u0 = bfbits(tanhf(c0[j] + xv0[j]));
            hnxt[r0 * DH_ + col] = u0;
            hs[((size_t)r0 * T_ + t) * DH_ + col] = u0;

            const int r1 = r0 + 16;                      // batch rows 16-31
            unsigned short u1 = bfbits(tanhf(c1[j] + xv1[j]));
            hnxt[r1 * DH_ + col] = u1;
            hs[((size_t)r1 * T_ + t) * DH_ + col] = u1;
        }
        __syncthreads();                                 // single barrier per step
        p ^= 1;
    }
}

// ---------------------------------------------------------------------------
// Readout: out = hs @ W_out + b_out.  One block per M-tile; each wave owns
// 2 N-tiles sharing one A-fragment stream (halves hs re-read traffic).
// ---------------------------------------------------------------------------
__global__ __launch_bounds__(256, 1) void k_outproj(const unsigned short* __restrict__ hs,     // [B*T][DH] bf16
                                                    const unsigned short* __restrict__ wt_out, // [DOUT][DH] bf16
                                                    const float* __restrict__ bout,
                                                    float* __restrict__ out) {
    const int tid  = threadIdx.x;
    const int lane = tid & 31, wave = tid >> 5;
    const int m0   = blockIdx.x << 4;                    // 8192 M-tiles
    const int lrow = lane & 15, half = lane >> 4;

    const unsigned short* arow = hs + (size_t)(m0 + lrow) * DH_;
    const unsigned short* brow[2];
    int n0[2];
#pragma unroll
    for (int q = 0; q < 2; ++q) {
        n0[q]   = ((wave << 1) + q) << 4;                // waves cover all 16 N-tiles
        brow[q] = wt_out + (size_t)(n0[q] + lrow) * DH_;
    }

    v8f c[2] = {};
    // 2-stage pipeline over K
    v16bf pa = load_a_bf16(arow, half * 16);
    v16bf pb[2];
#pragma unroll
    for (int q = 0; q < 2; ++q) pb[q] = *(const v16bf*)(brow[q] + half * 16);

#pragma unroll 4
    for (int kk = 0; kk < DH_; kk += 32) {
        const int kn = (kk + 32 < DH_) ? kk + 32 : kk;
        v16bf na = load_a_bf16(arow, kn + half * 16);
        v16bf nb[2];
#pragma unroll
        for (int q = 0; q < 2; ++q) nb[q] = *(const v16bf*)(brow[q] + kn + half * 16);

#pragma unroll
        for (int q = 0; q < 2; ++q)
            c[q] = __builtin_amdgcn_wmma_f32_16x16x32_bf16(false, pa, false, pb[q],
                                                           (short)0, c[q], false, false);
        pa = na;
#pragma unroll
        for (int q = 0; q < 2; ++q) pb[q] = nb[q];

        // per-iteration schedule: k+1 loads first, then k's WMMAs
        SCHED_GB(0x020, 6);   // 6 global_load_b128 (A + 2 B fragments)
        SCHED_GB(0x008, 2);   // 2 v_wmma
    }

#pragma unroll
    for (int q = 0; q < 2; ++q) {
        const int col  = n0[q] + lrow;
        const float bias = bout[col];
#pragma unroll
        for (int j = 0; j < 8; ++j) {
            const int m = m0 + half * 8 + j;
            out[(size_t)m * DOUT_ + col] = c[q][j] + bias;
        }
    }
}

// ---------------------------------------------------------------------------
// Host-side launch
// ---------------------------------------------------------------------------
extern "C" void kernel_launch(void* const* d_in, const int* in_sizes, int n_in,
                              void* d_out, int out_size, void* d_ws, size_t ws_size,
                              hipStream_t stream) {
    (void)in_sizes; (void)n_in; (void)out_size; (void)ws_size;

    const float* x     = (const float*)d_in[0];  // [B,T,DIN]
    const float* h0    = (const float*)d_in[1];  // [B,DH]
    const float* W_ih  = (const float*)d_in[2];  // [DIN,DH]
    const float* W_hh  = (const float*)d_in[3];  // [DH,DH]
    const float* b_h   = (const float*)d_in[4];  // [DH]
    const float* W_out = (const float*)d_in[5];  // [DH,DOUT]
    const float* b_out = (const float*)d_in[6];  // [DOUT]
    float* out = (float*)d_out;                  // [B,T,DOUT]

    // Workspace layout
    const size_t XW_BYTES = (size_t)T_ * DH_ * B_ * sizeof(float);          // 268 MB
    const size_t HS_BYTES = (size_t)B_ * T_ * DH_ * sizeof(unsigned short); // 134 MB
    char* ws = (char*)d_ws;
    float*          xw     = (float*)ws;                                    // [T][DH][B]
    unsigned short* hs     = (unsigned short*)(ws + XW_BYTES);              // [B][T][DH]
    unsigned short* wt_ih  = (unsigned short*)(ws + XW_BYTES + HS_BYTES);   // [DH][DIN]
    unsigned short* wt_hh  = wt_ih + (size_t)DH_ * DIN_;                    // [DH][DH]
    unsigned short* wt_out = wt_hh + (size_t)DH_ * DH_;                     // [DOUT][DH]

    // 1) transpose + convert weights to bf16, N-major
    {
        int n;
        n = DIN_ * DH_;
        k_transpose_bf16<<<(n + 255) / 256, 256, 0, stream>>>(W_ih, wt_ih, DIN_, DH_);
        n = DH_ * DH_;
        k_transpose_bf16<<<(n + 255) / 256, 256, 0, stream>>>(W_hh, wt_hh, DH_, DH_);
        n = DH_ * DOUT_;
        k_transpose_bf16<<<(n + 255) / 256, 256, 0, stream>>>(W_out, wt_out, DH_, DOUT_);
    }

    // 2) input projection: one block per M-tile, LDS-staged A
    k_inproj<<<(B_ * T_) / 16, 256, 0, stream>>>(x, wt_ih, b_h, xw);

    // 3) sequential recurrence: one persistent workgroup, ping-pong h in LDS
    k_scan<<<1, 1024, 0, stream>>>(h0, wt_hh, xw, hs);

    // 4) readout GEMM: one block per M-tile, 2 N-tiles per wave
    k_outproj<<<(B_ * T_) / 16, 256, 0, stream>>>(hs, wt_out, b_out, out);
}
